// UNetModel_36867999269068
// MI455X (gfx1250) — matrix operand order repaired and checked by reading, not compiled, not run on hardware
//
#include <hip/hip_runtime.h>
#include <hip/hip_bf16.h>
#include <math.h>

// ---------------------------------------------------------------------------
// MI455X (gfx1250) implementation. wave32, WMMA f32_16x16x32_f16, async-to-LDS
// tile staging (ASYNCcnt) for block-shared operands.
// Shapes: B=8, C=512, H=W=32 -> S=1024, M=B*S=8192, NH=8 (hd=64), multi hd=256.
// ---------------------------------------------------------------------------

typedef __attribute__((ext_vector_type(16))) _Float16 v16h;
typedef __attribute__((ext_vector_type(8)))  float    v8f;

#define BB   8
#define CC   512
#define SS   1024
#define MM   (BB * SS)

__device__ __forceinline__ int lane_id() { return threadIdx.x & 31; }

// Low 32 bits of a generic pointer to __shared__ = logical LDS byte offset
// (AS3->AS0 cast produces {aperture_hi32, lds_offset}; HW maps addr[31:0]).
__device__ __forceinline__ unsigned lds_addr32(const void* p) {
    return (unsigned)(uintptr_t)p;
}

// GLOBAL_LOAD_ASYNC_TO_LDS_B128 (CDNA5 ISA §10.7/§15.18, ASYNCcnt-tracked).
__device__ __forceinline__ void async_ld_b128(unsigned lds_off, const void* gaddr) {
    asm volatile("global_load_async_to_lds_b128 %0, %1, off"
                 :: "v"(lds_off), "v"(gaddr) : "memory");
}
__device__ __forceinline__ void wait_async_le(int n) {
    // n is always a compile-time constant at call sites
    if (n == 0)      asm volatile("s_wait_asynccnt 0" ::: "memory");
    else if (n == 2) asm volatile("s_wait_asynccnt 2" ::: "memory");
    else if (n == 4) asm volatile("s_wait_asynccnt 4" ::: "memory");
    else             asm volatile("s_wait_asynccnt 16" ::: "memory");
}

// ---- WMMA fragment helpers (contiguous k order, same in A and B) -------------------
// A (16x32 f16, MxK): lane row = lane%16; K = 16*(lane/16) + {0..15}
__device__ __forceinline__ v16h load_a_f16(const _Float16* __restrict__ base, int ld) {
    const int l   = lane_id();
    const int row = l & 15;
    const int kb  = (l >> 4) << 4;           // 0 or 16
    const _Float16* p = base + (size_t)row * ld + kb;
    v16h a;
#pragma unroll
    for (int e = 0; e < 16; ++e) a[e] = p[e];
    return a;
}

// B (32x16 f16, KxN) where source is "N-major" [N][K] (W[out][in] or K rows):
// B[k][n] = src[n][k]. lane col = lane%16, K = 16*(lane/16)+e -> contiguous 16.
__device__ __forceinline__ v16h load_bT_f16(const _Float16* __restrict__ base, int ld) {
    const int l  = lane_id();
    const int n  = l & 15;
    const int kb = (l >> 4) << 4;            // 0 or 16
    const _Float16* p = base + (size_t)n * ld + kb;
    v16h b;
#pragma unroll
    for (int e = 0; e < 16; ++e) b[e] = p[e];
    return b;
}

// B (32x16 f16, KxN) from a row-major [K][N] source (V tiles): strided gather.
__device__ __forceinline__ v16h load_bR_f16(const _Float16* __restrict__ base, int ld) {
    const int l  = lane_id();
    const int n  = l & 15;
    const int kb = (l >> 4) << 4;            // 0 or 16
    const _Float16* p = base + (size_t)kb * ld + n;
    v16h b;
#pragma unroll
    for (int e = 0; e < 16; ++e) b[e] = p[(size_t)e * ld];
    return b;
}

__device__ __forceinline__ v8f wmma16(v16h a, v16h b, v8f c) {
    return __builtin_amdgcn_wmma_f32_16x16x32_f16(false, a, false, b, (short)0, c,
                                                  false, false);
}

// reductions across the 16 lanes holding one C-row (xor masks 1,2,4,8 stay in group)
__device__ __forceinline__ float rowred_max(float v) {
#pragma unroll
    for (int m = 1; m <= 8; m <<= 1) v = fmaxf(v, __shfl_xor(v, m, 32));
    return v;
}
__device__ __forceinline__ float rowred_sum(float v) {
#pragma unroll
    for (int m = 1; m <= 8; m <<= 1) v += __shfl_xor(v, m, 32);
    return v;
}

// ---------------------------------------------------------------------------
// f32 -> f16 convert (weights)
// ---------------------------------------------------------------------------
__global__ void f32_to_f16_kernel(const float* __restrict__ src,
                                  _Float16* __restrict__ dst, int n) {
    int i = blockIdx.x * 256 + threadIdx.x;
    if (i < n) dst[i] = (_Float16)src[i];
}

// ---------------------------------------------------------------------------
// GroupNorm (32 groups of 16 ch x 1024 spatial) variants
// ---------------------------------------------------------------------------
__global__ __launch_bounds__(256) void gn_pre_kernel(
    const float* __restrict__ xin, const float* __restrict__ gamma,
    const float* __restrict__ beta, const float* __restrict__ pos,
    float* __restrict__ xout) {
    const int b = blockIdx.x >> 5, grp = blockIdx.x & 31;
    const size_t base = ((size_t)b * CC + grp * 16) * SS;
    float s = 0.f, ss = 0.f;
    for (int i = threadIdx.x; i < 16 * SS; i += 256) {
        float v = xin[base + i]; s += v; ss += v * v;
    }
    __shared__ float red0[256], red1[256];
    red0[threadIdx.x] = s; red1[threadIdx.x] = ss;
    __syncthreads();
    for (int st = 128; st > 0; st >>= 1) {
        if (threadIdx.x < st) { red0[threadIdx.x] += red0[threadIdx.x + st];
                                red1[threadIdx.x] += red1[threadIdx.x + st]; }
        __syncthreads();
    }
    const float mean = red0[0] * (1.f / 16384.f);
    const float rstd = rsqrtf(red1[0] * (1.f / 16384.f) - mean * mean + 1e-6f);
    for (int i = threadIdx.x; i < 16 * SS; i += 256) {
        const int c = grp * 16 + (i >> 10), sp = i & 1023;
        xout[base + i] = (xin[base + i] - mean) * rstd * gamma[c] + beta[c] +
                         pos[(size_t)c * SS + sp];
    }
}

// GN over x, result written transposed as f16 [B,S,C]
__global__ __launch_bounds__(256) void gn_sh_kernel(
    const float* __restrict__ xin, const float* __restrict__ gamma,
    const float* __restrict__ beta, _Float16* __restrict__ sh) {
    const int b = blockIdx.x >> 5, grp = blockIdx.x & 31;
    const size_t base = ((size_t)b * CC + grp * 16) * SS;
    float s = 0.f, ss = 0.f;
    for (int i = threadIdx.x; i < 16 * SS; i += 256) {
        float v = xin[base + i]; s += v; ss += v * v;
    }
    __shared__ float red0[256], red1[256];
    red0[threadIdx.x] = s; red1[threadIdx.x] = ss;
    __syncthreads();
    for (int st = 128; st > 0; st >>= 1) {
        if (threadIdx.x < st) { red0[threadIdx.x] += red0[threadIdx.x + st];
                                red1[threadIdx.x] += red1[threadIdx.x + st]; }
        __syncthreads();
    }
    const float mean = red0[0] * (1.f / 16384.f);
    const float rstd = rsqrtf(red1[0] * (1.f / 16384.f) - mean * mean + 1e-6f);
    for (int i = threadIdx.x; i < 16 * SS; i += 256) {
        const int c = grp * 16 + (i >> 10), sp = i & 1023;
        float v = (xin[base + i] - mean) * rstd * gamma[c] + beta[c];
        sh[((size_t)b * SS + sp) * CC + c] = (_Float16)v;
    }
}

// post-GN over [B,S,C] f32 input, write [B,C,S] + residual
__global__ __launch_bounds__(256) void gn_post_kernel(
    const float* __restrict__ yin, const float* __restrict__ gamma,
    const float* __restrict__ beta, const float* __restrict__ resid,
    float* __restrict__ out) {
    const int b = blockIdx.x >> 5, grp = blockIdx.x & 31;
    float s = 0.f, ss = 0.f;
    for (int i = threadIdx.x; i < 16 * SS; i += 256) {
        const int c = grp * 16 + (i >> 10), sp = i & 1023;
        float v = yin[((size_t)b * SS + sp) * CC + c];
        s += v; ss += v * v;
    }
    __shared__ float red0[256], red1[256];
    red0[threadIdx.x] = s; red1[threadIdx.x] = ss;
    __syncthreads();
    for (int st = 128; st > 0; st >>= 1) {
        if (threadIdx.x < st) { red0[threadIdx.x] += red0[threadIdx.x + st];
                                red1[threadIdx.x] += red1[threadIdx.x + st]; }
        __syncthreads();
    }
    const float mean = red0[0] * (1.f / 16384.f);
    const float rstd = rsqrtf(red1[0] * (1.f / 16384.f) - mean * mean + 1e-6f);
    for (int i = threadIdx.x; i < 16 * SS; i += 256) {
        const int c = grp * 16 + (i >> 10), sp = i & 1023;
        float v = yin[((size_t)b * SS + sp) * CC + c];
        float r = (v - mean) * rstd * gamma[c] + beta[c];
        const size_t oidx = ((size_t)b * CC + c) * SS + sp;
        out[oidx] = r + resid[oidx];
    }
}

// ---------------------------------------------------------------------------
// Spatial gate convs (tiny: ~0.8 GF total)
// ---------------------------------------------------------------------------
__global__ void conv1_silu_kernel(const float* __restrict__ x,
                                  const float* __restrict__ w,
                                  const float* __restrict__ bias,
                                  float* __restrict__ out) {
    const int idx = blockIdx.x * 256 + threadIdx.x;       // B*64*1024
    const int sp = idx & 1023, o = (idx >> 10) & 63, b = idx >> 16;
    const float* xb = x + (size_t)b * CC * SS + sp;
    const float* wo = w + o * CC;
    float acc = bias[o];
    for (int c = 0; c < CC; ++c) acc += xb[(size_t)c * SS] * wo[c];
    out[idx] = acc / (1.f + __expf(-acc));
}

__global__ void conv3x3_silu_kernel(const float* __restrict__ x,
                                    const float* __restrict__ w,
                                    const float* __restrict__ bias,
                                    float* __restrict__ out) {
    const int idx = blockIdx.x * 256 + threadIdx.x;       // B*64*1024
    const int sp = idx & 1023, o = (idx >> 10) & 63, b = idx >> 16;
    const int yy = sp >> 5, xx = sp & 31;
    float acc = bias[o];
    for (int ci = 0; ci < 64; ++ci) {
        const float* xc = x + (((size_t)b * 64 + ci) << 10);
        const float* wc = w + ((o * 64 + ci) * 9);
        for (int ky = 0; ky < 3; ++ky) {
            const int iy = yy + ky - 1;
            if (iy < 0 || iy > 31) continue;
            for (int kx = 0; kx < 3; ++kx) {
                const int ix = xx + kx - 1;
                if (ix < 0 || ix > 31) continue;
                acc += xc[iy * 32 + ix] * wc[ky * 3 + kx];
            }
        }
    }
    out[idx] = acc / (1.f + __expf(-acc));
}

__global__ void conv1_sigmoid_kernel(const float* __restrict__ x,
                                     const float* __restrict__ w,
                                     const float* __restrict__ bias,
                                     float* __restrict__ out) {
    const int idx = blockIdx.x * 256 + threadIdx.x;       // B*1024
    const int sp = idx & 1023, b = idx >> 10;
    float acc = bias[0];
    for (int ci = 0; ci < 64; ++ci)
        acc += x[(((size_t)b * 64 + ci) << 10) + sp] * w[ci];
    out[idx] = 1.f / (1.f + __expf(-acc));
}

// msh[b][s][c] = f16( x[b][c][s] * sw[b][s] )
__global__ void gate_transpose_kernel(const float* __restrict__ x,
                                      const float* __restrict__ sw,
                                      _Float16* __restrict__ msh) {
    const size_t idx = (size_t)blockIdx.x * 256 + threadIdx.x;   // B*S*C
    const int c = idx & (CC - 1);
    const size_t bs = idx >> 9;
    const int sp = bs & (SS - 1), b = (int)(bs >> 10);
    msh[idx] = (_Float16)(x[((size_t)b * CC + c) * SS + sp] * sw[bs]);
}

// ---------------------------------------------------------------------------
// WMMA GEMM: Y = act(X * W^T + bias).  X:[M,K] f16, W:[N,K] f16.
// Block tile 64x64: 4 waves, each wave a 16x64 strip (4 independent
// accumulators -> no WMMA->WMMA RAW stalls).  The W tile (shared by all 4
// waves) is staged to LDS with double-buffered GLOBAL_LOAD_ASYNC_TO_LDS_B128.
// ---------------------------------------------------------------------------
template <int ACT, int BIAS, int OUTF16>
__global__ __launch_bounds__(128) void gemm_xwt_kernel(
    const _Float16* __restrict__ X, const _Float16* __restrict__ W,
    const float* __restrict__ bias, void* __restrict__ Yv,
    int M, int N, int K, int ldy, int ycol0) {
    __shared__ _Float16 wtile[2][64 * 32];   // [buf][n*32 + k]

    const int wave = threadIdx.x >> 5;
    const int l = lane_id();
    const int tid = threadIdx.x;
    const int row0 = blockIdx.x * 64 + wave * 16;
    const int col0 = blockIdx.y * 64;

    // stage W[col0..col0+63][kk..kk+31] -> wtile[buf]; 2 x b128 per thread
    auto issue_w = [&](int buf, int kk) {
#pragma unroll
        for (int c = 0; c < 2; ++c) {
            const int chunk = tid * 2 + c;          // 0..255
            const int n = chunk >> 2, part = (chunk & 3) * 8;
            async_ld_b128(lds_addr32(&wtile[buf][n * 32 + part]),
                          W + (size_t)(col0 + n) * K + kk + part);
        }
    };

    v8f acc[4];
#pragma unroll
    for (int t = 0; t < 4; ++t)
        acc[t] = (v8f){0.f, 0.f, 0.f, 0.f, 0.f, 0.f, 0.f, 0.f};

    const _Float16* xp = X + (size_t)row0 * K;
    issue_w(0, 0);
    for (int k = 0, it = 0; k < K; k += 32, ++it) {
        __syncthreads();                         // prior reads done before overwrite
        if (k + 32 < K) { issue_w((it + 1) & 1, k + 32); wait_async_le(2); }
        else            { wait_async_le(0); }
        __syncthreads();                         // staged tile visible to all waves
        const _Float16* wb = &wtile[it & 1][0];
        const v16h a = load_a_f16(xp + k, K);
#pragma unroll
        for (int t = 0; t < 4; ++t)
            acc[t] = wmma16(a, load_bT_f16(wb + (size_t)(t * 16) * 32, 32), acc[t]);
    }

    const int rbase = (l >> 4) << 3;
    const int col = l & 15;
#pragma unroll
    for (int t = 0; t < 4; ++t) {
        const int gcol = col0 + t * 16 + col;
        const float bv = BIAS ? bias[gcol] : 0.f;
#pragma unroll
        for (int v = 0; v < 8; ++v) {
            float y = acc[t][v] + bv;
            if (ACT == 1) y = y / (1.f + __expf(-y));     // silu
            const size_t idx = (size_t)(row0 + rbase + v) * ldy + ycol0 + gcol;
            if (OUTF16) ((_Float16*)Yv)[idx] = (_Float16)y;
            else        ((float*)Yv)[idx] = y;
        }
    }
}

// ---------------------------------------------------------------------------
// Flash attention (online softmax), WMMA for QK^T and P*V.
// Q/K/V: [B, S, nheads*HD] f16 row-major (ld = ldq).  One wave = 16 queries;
// the 4 waves of a block share (b, head), so the 32-key K/V tiles are staged
// once per block via double-buffered async-to-LDS.  Probs go through a 1KB
// LDS stage per wave to turn the C-layout into an A-fragment for PV.
// Output written f16 into combined[M, ldo] at column (ocol0 + head*HD).
// ---------------------------------------------------------------------------
template <int HD>
__global__ __launch_bounds__(128) void flash_attn_kernel(
    const _Float16* __restrict__ Qb, const _Float16* __restrict__ Kb,
    const _Float16* __restrict__ Vb, _Float16* __restrict__ Out,
    int ldq, int ldo, int ocol0, float scale) {
    constexpr int KA = HD / 32;     // A fragments per 16-row Q tile
    constexpr int OT = HD / 16;     // 16-col output tiles
    constexpr int PARTS = HD / 8;   // 16-byte chunks per staged row
    constexpr int CPT = HD / 16;    // async ops per thread per (K+V) tile pair

    __shared__ _Float16 ktile[2][32 * HD];
    __shared__ _Float16 vtile[2][32 * HD];
    __shared__ _Float16 pstage[4][16][32];

    const int wave = threadIdx.x >> 5;
    const int l = lane_id();
    const int tid = threadIdx.x;
    const int b = blockIdx.z, h = blockIdx.y;
    const int q0 = blockIdx.x * 64 + wave * 16;

    const _Float16* Qp = Qb + (size_t)b * SS * ldq + h * HD;
    const _Float16* Kp = Kb + (size_t)b * SS * ldq + h * HD;
    const _Float16* Vp = Vb + (size_t)b * SS * ldq + h * HD;

    auto issue_kv = [&](int buf, int j) {
        for (int c = tid; c < 32 * PARTS; c += 128) {
            const int r = c / PARTS, part = (c % PARTS) * 8;
            async_ld_b128(lds_addr32(&ktile[buf][r * HD + part]),
                          Kp + (size_t)(j + r) * ldq + part);
            async_ld_b128(lds_addr32(&vtile[buf][r * HD + part]),
                          Vp + (size_t)(j + r) * ldq + part);
        }
    };

    v16h aq[KA];
#pragma unroll
    for (int t = 0; t < KA; ++t)
        aq[t] = load_a_f16(Qp + (size_t)q0 * ldq + t * 32, ldq);

    v8f o[OT];
#pragma unroll
    for (int t = 0; t < OT; ++t)
        o[t] = (v8f){0.f, 0.f, 0.f, 0.f, 0.f, 0.f, 0.f, 0.f};
    float m[8], lsum[8];
#pragma unroll
    for (int v = 0; v < 8; ++v) { m[v] = -1e30f; lsum[v] = 0.f; }

    const int rbase = (l >> 4) << 3;
    const int col = l & 15;

    issue_kv(0, 0);
    for (int j = 0, it = 0; j < SS; j += 32, ++it) {
        __syncthreads();                    // prior buffer reads done before overwrite
        if (j + 32 < SS) { issue_kv((it + 1) & 1, j + 32); wait_async_le(CPT); }
        else             { wait_async_le(0); }
        __syncthreads();                    // staged K/V visible to all waves
        const _Float16* kb = &ktile[it & 1][0];
        const _Float16* vb = &vtile[it & 1][0];

        // ---- scores for 32 keys (two 16-key tiles) ----
        v8f s0 = {0.f, 0.f, 0.f, 0.f, 0.f, 0.f, 0.f, 0.f};
        v8f s1 = {0.f, 0.f, 0.f, 0.f, 0.f, 0.f, 0.f, 0.f};
#pragma unroll
        for (int t = 0; t < KA; ++t) {
            s0 = wmma16(aq[t], load_bT_f16(kb + t * 32, HD), s0);
            s1 = wmma16(aq[t], load_bT_f16(kb + 16 * HD + t * 32, HD), s1);
        }
        // ---- online softmax update ----
        float p0[8], p1[8], corr[8];
#pragma unroll
        for (int v = 0; v < 8; ++v) {
            const float x0 = s0[v] * scale, x1 = s1[v] * scale;
            const float mx = rowred_max(fmaxf(x0, x1));
            const float mn = fmaxf(m[v], mx);
            corr[v] = __expf(m[v] - mn);
            p0[v] = __expf(x0 - mn);
            p1[v] = __expf(x1 - mn);
            lsum[v] = lsum[v] * corr[v] + rowred_sum(p0[v] + p1[v]);
            m[v] = mn;
        }
#pragma unroll
        for (int t = 0; t < OT; ++t)
#pragma unroll
            for (int v = 0; v < 8; ++v) o[t][v] *= corr[v];

        // ---- redistribute probs: C layout -> A fragment via LDS ----
        __syncthreads();
#pragma unroll
        for (int v = 0; v < 8; ++v) {
            pstage[wave][rbase + v][col]      = (_Float16)p0[v];
            pstage[wave][rbase + v][16 + col] = (_Float16)p1[v];
        }
        __syncthreads();
        const v16h ap = load_a_f16(&pstage[wave][0][0], 32);

        // ---- accumulate P * V ----
#pragma unroll
        for (int t = 0; t < OT; ++t)
            o[t] = wmma16(ap, load_bR_f16(vb + t * 16, HD), o[t]);
    }

    // ---- normalize and store ----
#pragma unroll
    for (int t = 0; t < OT; ++t)
#pragma unroll
        for (int v = 0; v < 8; ++v) {
            const float val = o[t][v] / lsum[v];
            const size_t row = (size_t)b * SS + q0 + rbase + v;
            Out[row * ldo + ocol0 + h * HD + t * 16 + col] = (_Float16)val;
        }
}

// ---------------------------------------------------------------------------
// Host orchestration
// ---------------------------------------------------------------------------
extern "C" void kernel_launch(void* const* d_in, const int* in_sizes, int n_in,
                              void* d_out, int out_size, void* d_ws, size_t ws_size,
                              hipStream_t stream) {
    (void)in_sizes; (void)n_in; (void)out_size; (void)ws_size;

    const float* hidden = (const float*)d_in[0];
    const float* pre_g = (const float*)d_in[1];
    const float* pre_b = (const float*)d_in[2];
    const float* norm_g = (const float*)d_in[3];
    const float* norm_b = (const float*)d_in[4];
    const float* post_g = (const float*)d_in[5];
    const float* post_b = (const float*)d_in[6];
    const float* pos = (const float*)d_in[7];
    const float* wq = (const float*)d_in[8];
    const float* wk = (const float*)d_in[9];
    const float* wv = (const float*)d_in[10];
    const float* wm[6] = {(const float*)d_in[11], (const float*)d_in[12],
                          (const float*)d_in[13], (const float*)d_in[14],
                          (const float*)d_in[15], (const float*)d_in[16]};
    const float* sa_w1 = (const float*)d_in[17];
    const float* sa_b1 = (const float*)d_in[18];
    const float* sa_w2 = (const float*)d_in[19];
    const float* sa_b2 = (const float*)d_in[20];
    const float* sa_w3 = (const float*)d_in[21];
    const float* sa_b3 = (const float*)d_in[22];
    const float* ff_w1 = (const float*)d_in[23];
    const float* ff_b1 = (const float*)d_in[24];
    const float* ff_w2 = (const float*)d_in[25];
    const float* ff_b2 = (const float*)d_in[26];
    const float* out_w = (const float*)d_in[27];
    const float* out_b = (const float*)d_in[28];
    float* out = (float*)d_out;

    // ---- workspace carve-up ----
    size_t off = 0;
    auto alloc = [&](size_t bytes) {
        void* p = (char*)d_ws + off;
        off = (off + bytes + 255) & ~(size_t)255;
        return p;
    };
    float* x      = (float*)alloc(sizeof(float) * (size_t)BB * CC * SS);
    float* g1     = (float*)alloc(sizeof(float) * (size_t)BB * 64 * SS);
    float* g2     = (float*)alloc(sizeof(float) * (size_t)BB * 64 * SS);
    float* swg    = (float*)alloc(sizeof(float) * (size_t)BB * SS);
    _Float16* sh16  = (_Float16*)alloc(2ull * BB * SS * CC);
    _Float16* msh16 = (_Float16*)alloc(2ull * BB * SS * CC);
    _Float16* wq16  = (_Float16*)alloc(2ull * CC * CC);
    _Float16* wk16  = (_Float16*)alloc(2ull * CC * CC);
    _Float16* wv16  = (_Float16*)alloc(2ull * CC * CC);
    _Float16* wm16[6];
    for (int i = 0; i < 6; ++i) wm16[i] = (_Float16*)alloc(2ull * 256 * CC);
    _Float16* ffw1_16 = (_Float16*)alloc(2ull * CC * 1024);
    _Float16* ffw2_16 = (_Float16*)alloc(2ull * CC * CC);
    _Float16* outw16  = (_Float16*)alloc(2ull * CC * CC);
    _Float16* q16 = (_Float16*)alloc(2ull * MM * CC);
    _Float16* k16 = (_Float16*)alloc(2ull * MM * CC);
    _Float16* v16 = (_Float16*)alloc(2ull * MM * CC);
    _Float16* qm16[2], *km16[2], *vm16[2];
    for (int i = 0; i < 2; ++i) {
        qm16[i] = (_Float16*)alloc(2ull * MM * 256);
        km16[i] = (_Float16*)alloc(2ull * MM * 256);
        vm16[i] = (_Float16*)alloc(2ull * MM * 256);
    }
    _Float16* comb16 = (_Float16*)alloc(2ull * MM * 1024);   // [std | ms0 | ms1]
    _Float16* ff1_16 = (_Float16*)alloc(2ull * MM * CC);
    _Float16* ff2_16 = (_Float16*)alloc(2ull * MM * CC);
    float* finalf    = (float*)alloc(sizeof(float) * (size_t)MM * CC);

    // ---- weights -> f16 ----
    auto cvt = [&](const float* s, _Float16* d, int n) {
        f32_to_f16_kernel<<<(n + 255) / 256, 256, 0, stream>>>(s, d, n);
    };
    cvt(wq, wq16, CC * CC); cvt(wk, wk16, CC * CC); cvt(wv, wv16, CC * CC);
    for (int i = 0; i < 6; ++i) cvt(wm[i], wm16[i], 256 * CC);
    cvt(ff_w1, ffw1_16, CC * 1024);
    cvt(ff_w2, ffw2_16, CC * CC);
    cvt(out_w, outw16, CC * CC);

    // ---- pre GN (+pos), sh = GN(x) transposed f16 ----
    gn_pre_kernel<<<BB * 32, 256, 0, stream>>>(hidden, pre_g, pre_b, pos, x);
    gn_sh_kernel<<<BB * 32, 256, 0, stream>>>(x, norm_g, norm_b, sh16);

    // ---- spatial gate ----
    conv1_silu_kernel<<<(BB * 64 * SS) / 256, 256, 0, stream>>>(x, sa_w1, sa_b1, g1);
    conv3x3_silu_kernel<<<(BB * 64 * SS) / 256, 256, 0, stream>>>(g1, sa_w2, sa_b2, g2);
    conv1_sigmoid_kernel<<<(BB * SS) / 256, 256, 0, stream>>>(g2, sa_w3, sa_b3, swg);
    gate_transpose_kernel<<<(int)(((size_t)BB * SS * CC) / 256), 256, 0, stream>>>(
        x, swg, msh16);

    // ---- projections (WMMA GEMM, Y = X*W^T), 64x64 block tiles ----
    const dim3 gemm_blk(128);
    gemm_xwt_kernel<0, 0, 1><<<dim3(MM / 64, CC / 64), gemm_blk, 0, stream>>>(
        sh16, wq16, nullptr, q16, MM, CC, CC, CC, 0);
    gemm_xwt_kernel<0, 0, 1><<<dim3(MM / 64, CC / 64), gemm_blk, 0, stream>>>(
        sh16, wk16, nullptr, k16, MM, CC, CC, CC, 0);
    gemm_xwt_kernel<0, 0, 1><<<dim3(MM / 64, CC / 64), gemm_blk, 0, stream>>>(
        sh16, wv16, nullptr, v16, MM, CC, CC, CC, 0);
    for (int p = 0; p < 2; ++p) {
        gemm_xwt_kernel<0, 0, 1><<<dim3(MM / 64, 256 / 64), gemm_blk, 0, stream>>>(
            msh16, wm16[p], nullptr, qm16[p], MM, 256, CC, 256, 0);       // wq0/wq1
        gemm_xwt_kernel<0, 0, 1><<<dim3(MM / 64, 256 / 64), gemm_blk, 0, stream>>>(
            sh16, wm16[2 + p], nullptr, km16[p], MM, 256, CC, 256, 0);    // wk0/wk1
        gemm_xwt_kernel<0, 0, 1><<<dim3(MM / 64, 256 / 64), gemm_blk, 0, stream>>>(
            sh16, wm16[4 + p], nullptr, vm16[p], MM, 256, CC, 256, 0);    // wv0/wv1
    }

    // ---- attention: standard (8 heads, hd=64) + two multi-scale (hd=256) ----
    flash_attn_kernel<64><<<dim3(SS / 64, 8, BB), gemm_blk, 0, stream>>>(
        q16, k16, v16, comb16, CC, 1024, 0, 0.125f);                 // 1/sqrt(64)
    flash_attn_kernel<256><<<dim3(SS / 64, 1, BB), gemm_blk, 0, stream>>>(
        qm16[0], km16[0], vm16[0], comb16, 256, 1024, 512, 0.0625f); // 1/sqrt(256)
    flash_attn_kernel<256><<<dim3(SS / 64, 1, BB), gemm_blk, 0, stream>>>(
        qm16[1], km16[1], vm16[1], comb16, 256, 1024, 768, 0.0625f);

    // ---- FF: silu(comb @ W1^T + b1) @ W2^T + b2, then out-proj ----
    gemm_xwt_kernel<1, 1, 1><<<dim3(MM / 64, CC / 64), gemm_blk, 0, stream>>>(
        comb16, ffw1_16, ff_b1, ff1_16, MM, CC, 1024, CC, 0);
    gemm_xwt_kernel<0, 1, 1><<<dim3(MM / 64, CC / 64), gemm_blk, 0, stream>>>(
        ff1_16, ffw2_16, ff_b2, ff2_16, MM, CC, CC, CC, 0);
    gemm_xwt_kernel<0, 1, 0><<<dim3(MM / 64, CC / 64), gemm_blk, 0, stream>>>(
        ff2_16, outw16, out_b, finalf, MM, CC, CC, CC, 0);

    // ---- post GN + residual, transpose back to [B,C,H,W] ----
    gn_post_kernel<<<BB * 32, 256, 0, stream>>>(finalf, post_g, post_b, hidden, out);
}
